// BaselineModel_83176336654929
// MI455X (gfx1250) — matrix operand order repaired and checked by reading, not compiled, not run on hardware
//
#include <hip/hip_runtime.h>
#include <hip/hip_bf16.h>
#include <math.h>

// ---------------------------------------------------------------------------
// Types for CDNA5 WMMA (wave32): bf16 A/B fragments (16 halfs), f32 C/D (8 f32)
// ---------------------------------------------------------------------------
typedef __bf16 bf16_t;
typedef __attribute__((ext_vector_type(16))) __bf16 v16bf;
typedef __attribute__((ext_vector_type(8)))  float  v8f;

#define DEVINL __device__ __forceinline__

// Problem dims
constexpr int NB = 32, NS = 512, NBS = NB * NS;   // batch, seq, rows
constexpr int NE = 300, NEp = 320;                // embed dim, padded to 32
constexpr int NH = 256, NG = 4 * NH;              // hidden, gates (1024)
constexpr int ND = 512;                           // 2*H model dim
constexpr int NT = 9, NTp = 16;                   // tags, padded

DEVINL bf16_t f2bf(float f) {
  union { float f; unsigned int u; } a; a.f = f;
  unsigned int r = a.u + 0x7FFFu + ((a.u >> 16) & 1u);
  union { unsigned short s; bf16_t b; } o; o.s = (unsigned short)(r >> 16);
  return o.b;
}

DEVINL float sigf(float x) { return 1.0f / (1.0f + __expf(-x)); }

// A fragment: 16x32 bf16 tile at (m0,k0) of row-major A (lda elements).
// Layout: lanes 0-15 -> K {0..15}; lanes 16-31 -> K {16..31} (matches B);
// each lane's 8-K runs are contiguous in memory -> two b128 loads.
DEVINL v16bf load_a_frag(const bf16_t* A, size_t lda, int m0, int k0, int lane) {
  const int row = m0 + (lane & 15);
  const int klo = k0 + ((lane & 16) ? 16 : 0);
  const int khi = klo + 8;
  union { v16bf v; uint4 q[2]; } r;
  const bf16_t* base = A + (size_t)row * lda;
  r.q[0] = *(const uint4*)(base + klo);   // K pairs (klo+0,1)..(klo+6,7)
  r.q[1] = *(const uint4*)(base + khi);   // K pairs (khi+0,1)..(khi+6,7)
  return r.v;
}

// B fragment, B stored [N,K] row-major (computes A*B^T).
// Lanes 0-15: K=k0..k0+15 of column (lane&15), lanes 16-31: K=k0+16..k0+31.
// 32 contiguous bytes per lane -> two b128 loads.
DEVINL v16bf load_b_frag_nk(const bf16_t* Bm, size_t ldb, int n0, int k0, int lane) {
  const int col = n0 + (lane & 15);
  const int kk  = k0 + ((lane & 16) ? 16 : 0);
  union { v16bf v; uint4 q[2]; } r;
  const uint4* p = (const uint4*)(Bm + (size_t)col * ldb + kk);
  r.q[0] = p[0];
  r.q[1] = p[1];
  return r.v;
}

// B fragment, B stored [K,N] row-major (computes A*B).
DEVINL v16bf load_b_frag_kn(const bf16_t* Bm, size_t ldb, int n0, int k0, int lane) {
  const int col = n0 + (lane & 15);
  const int kb  = k0 + ((lane & 16) ? 16 : 0);
  union { v16bf v; unsigned short h[16]; } r;
  const unsigned short* p = (const unsigned short*)Bm;
#pragma unroll
  for (int e = 0; e < 16; ++e) r.h[e] = p[(size_t)(kb + e) * ldb + col];
  return r.v;
}

DEVINL v8f wmma_bf16(v16bf a, v16bf b, v8f c) {
  // (neg_a, A, neg_b, B, c_mod, C, reuse_a, reuse_b)
  return __builtin_amdgcn_wmma_f32_16x16x32_bf16(false, a, false, b, (short)0, c,
                                                 false, false);
}

// ---------------------------------------------------------------------------
// Generic tiled GEMM: C[M,N] = scale * A[M,K] (*) B (+ bias[col])
//   B_IS_KN == 0 : B stored [N,K] row-major  (A*B^T)
//   B_IS_KN == 1 : B stored [K,N] row-major  (A*B)
// block = 64 threads (2 waves); each wave computes a 16x64 strip (4 N tiles,
// A fragment reused 4x). grid (ceil(N/64), M/32, Z). N-tile count is
// wave-uniform so EXEC stays all-ones for every WMMA.
// ---------------------------------------------------------------------------
template <int B_IS_KN>
__global__ void gemm_kernel(const bf16_t* __restrict__ A, const bf16_t* __restrict__ Bm,
                            float* Cf, bf16_t* Cbf, const float* bias,
                            int M, int N, int K, int lda, int ldb, int ldc,
                            long long sA, long long sB, long long sC, float scale) {
  const int z = blockIdx.z;
  const bf16_t* Az = A + (size_t)z * sA;
  const bf16_t* Bz = Bm + (size_t)z * sB;
  const int lane  = threadIdx.x & 31;
  const int wave  = threadIdx.x >> 5;
  const int m0    = blockIdx.y * 32 + wave * 16;
  const int nbase = blockIdx.x * 64;
  if (m0 >= M) return;
  const int ntiles = (N - nbase) >= 64 ? 4 : (N - nbase) / 16;

  v8f acc[4] = {};
  for (int k0 = 0; k0 < K; k0 += 32) {
    v16bf a = load_a_frag(Az, (size_t)lda, m0, k0, lane);
    if (k0 + 32 < K)  // CDNA5 global_prefetch_b8 for next K slab of A
      __builtin_prefetch(Az + (size_t)(m0 + (lane & 15)) * lda + k0 + 32, 0, 1);
    for (int j = 0; j < ntiles; ++j) {
      const int n0 = nbase + j * 16;
      v16bf b = B_IS_KN ? load_b_frag_kn(Bz, (size_t)ldb, n0, k0, lane)
                        : load_b_frag_nk(Bz, (size_t)ldb, n0, k0, lane);
      acc[j] = wmma_bf16(a, b, acc[j]);
    }
  }

  const int rb = m0 + ((lane & 16) ? 8 : 0);
  for (int j = 0; j < ntiles; ++j) {
    const int col = nbase + j * 16 + (lane & 15);
    const float bs = bias ? bias[col] : 0.0f;
    if (Cf) {
      float* C = Cf + (size_t)z * sC;
#pragma unroll
      for (int r = 0; r < 8; ++r) C[(size_t)(rb + r) * ldc + col] = acc[j][r] * scale + bs;
    }
    if (Cbf) {
      bf16_t* C = Cbf + (size_t)z * sC;
#pragma unroll
      for (int r = 0; r < 8; ++r) C[(size_t)(rb + r) * ldc + col] = f2bf(acc[j][r] * scale + bs);
    }
  }
}

// ---------------------------------------------------------------------------
// Cooperative copy of 64 KB of weights (2048 cols x 16 K bf16, chunk c -> col c>>1)
// from global w_hh[col][k0+*] into an LDS slab laid out [col][32].
// Two phases: stage all 8 b128 chunks into registers (one load clause, one
// latency), then ds_store_b128 them. Avoids per-chunk load->wait->store
// round-trip serialization.
// ---------------------------------------------------------------------------
DEVINL void copy_whh_slab(const bf16_t* __restrict__ whh, int k0,
                          bf16_t* __restrict__ dst, int tid) {
  uint4 tmp[8];
#pragma unroll
  for (int i = 0; i < 8; ++i) {
    const int c   = tid + i * 512;   // 16-byte chunk id, 0..4095
    const int col = c >> 1;
    const int off = (c & 1) * 8;
    tmp[i] = *(const uint4*)(whh + (size_t)col * NH + k0 + off);
  }
#pragma unroll
  for (int i = 0; i < 8; ++i) {
    const int c   = tid + i * 512;
    const int col = c >> 1;
    const int off = (c & 1) * 8;
    *(uint4*)(dst + col * 32 + off) = tmp[i];
  }
}

// ---------------------------------------------------------------------------
// Persistent LSTM direction scan. grid.x = 2 (dir), block = 512 (16 waves).
// h (bf16) / c (f32) live in LDS; per step: [32,1024] = h[32,256] @ w_hh^T via
// WMMA (each wave: 1 M-tile x 8 N-tiles, K-loop of 8), then fused gates.
// w_hh is streamed through double-buffered LDS slabs so B fragments come from
// ds_load_b128 (no register spills, global latency hidden behind WMMAs).
// Dynamic LDS: 16KB h + 32KB c + 128KB gates + 2x64KB weight slabs = 304KB.
// ---------------------------------------------------------------------------
__global__ void __launch_bounds__(512)
lstm_scan_kernel(const float* __restrict__ xg0, const float* __restrict__ xg1,
                 const bf16_t* __restrict__ whh0, const bf16_t* __restrict__ whh1,
                 const float* __restrict__ bias0, const float* __restrict__ bias1,
                 bf16_t* __restrict__ out /* [NBS, 512] */) {
  const int dir = blockIdx.x;
  const float*  xg   = dir ? xg1 : xg0;
  const bf16_t* whh  = dir ? whh1 : whh0;
  const float*  bias = dir ? bias1 : bias0;

  extern __shared__ char smem[];
  bf16_t* hS = (bf16_t*)smem;                       // [32][256] bf16  (16 KB)
  float*  cS = (float*)(smem + 16 * 1024);          // [32][256] f32   (32 KB)
  float*  gS = (float*)(smem + 48 * 1024);          // [32][1024] f32  (128 KB)
  bf16_t* wbuf0 = (bf16_t*)(smem + 176 * 1024);     // slab buf 0      (64 KB)
  bf16_t* wbuf1 = (bf16_t*)(smem + 240 * 1024);     // slab buf 1      (64 KB)

  const int tid = threadIdx.x;
  for (int i = tid; i < 32 * 256; i += 512) { hS[i] = f2bf(0.0f); cS[i] = 0.0f; }
  copy_whh_slab(whh, 0, wbuf0, tid);  // prime k=0 slab
  __syncthreads();

  const int lane = tid & 31;
  const int wave = tid >> 5;
  const int m0   = (wave & 1) * 16;   // rows 0-15 / 16-31
  const int ngrp = wave >> 1;         // 0..7 -> N tiles ngrp*8 .. +7

  for (int t = 0; t < 512; ++t) {
    v8f acc[8] = {};
    for (int kidx = 0; kidx < 8; ++kidx) {
      const int k0 = kidx * 32;
      bf16_t* cur = (kidx & 1) ? wbuf1 : wbuf0;
      bf16_t* nxt = (kidx & 1) ? wbuf0 : wbuf1;
      v16bf a = load_a_frag(hS, 256, m0, k0, lane);
#pragma unroll
      for (int j = 0; j < 8; ++j) {
        const int n0 = (ngrp * 8 + j) * 16;
        v16bf b = load_b_frag_nk(cur, 32, n0, 0, lane);  // ds_load_b128 x2
        acc[j] = wmma_bf16(a, b, acc[j]);
      }
      // stream next slab (k+32, wrapping to k=0 for the next timestep)
      const int knext = (kidx < 7) ? (k0 + 32) : 0;
      copy_whh_slab(whh, knext, nxt, tid);
      __syncthreads();
    }
    // scatter D tiles to LDS gate scratch
    const int col = lane & 15;
    const int rb  = m0 + ((lane & 16) ? 8 : 0);
#pragma unroll
    for (int j = 0; j < 8; ++j) {
      const int n0 = (ngrp * 8 + j) * 16;
#pragma unroll
      for (int r = 0; r < 8; ++r) gS[(size_t)(rb + r) * 1024 + n0 + col] = acc[j][r];
    }
    __syncthreads();

    const int te = dir ? (511 - t) : t;
    for (int u = tid; u < 32 * 256; u += 512) {
      const int b = u >> 8;
      const int j = u & 255;
      const size_t row = (size_t)b * 512 + te;
      const float* xr = xg + row * 1024;
      const float gi = gS[b * 1024 + j]       + xr[j]       + bias[j];
      const float gf = gS[b * 1024 + 256 + j] + xr[256 + j] + bias[256 + j];
      const float gg = gS[b * 1024 + 512 + j] + xr[512 + j] + bias[512 + j];
      const float go = gS[b * 1024 + 768 + j] + xr[768 + j] + bias[768 + j];
      const float c  = sigf(gf) * cS[u] + sigf(gi) * tanhf(gg);
      const float h  = sigf(go) * tanhf(c);
      cS[u] = c;
      hS[u] = f2bf(h);
      out[row * 512 + dir * 256 + j] = f2bf(h);
    }
    __syncthreads();
  }
}

// ---------------------------------------------------------------------------
// Small helper kernels
// ---------------------------------------------------------------------------
__global__ void embed_gather_kernel(const int* __restrict__ ids,
                                    const float* __restrict__ table,
                                    bf16_t* __restrict__ x0) {
  const int row = blockIdx.x;
  const int id  = ids[row];
  const int c   = threadIdx.x;  // 0..319
  const float v = (c < NE) ? table[(size_t)id * NE + c] : 0.0f;
  x0[(size_t)row * NEp + c] = f2bf(v);
}

__global__ void pack_pad_kernel(const float* __restrict__ src, bf16_t* __restrict__ dst,
                                int srows, int scols, int drows, int dcols) {
  const int idx = blockIdx.x * blockDim.x + threadIdx.x;
  if (idx >= drows * dcols) return;
  const int r = idx / dcols, c = idx % dcols;
  const float v = (r < srows && c < scols) ? src[(size_t)r * scols + c] : 0.0f;
  dst[idx] = f2bf(v);
}

__global__ void add2_kernel(const float* a, const float* b, float* o, int n) {
  const int i = blockIdx.x * blockDim.x + threadIdx.x;
  if (i < n) o[i] = a[i] + b[i];
}

__global__ void pad_vec_kernel(const float* a, float* o, int n, int np) {
  const int i = threadIdx.x;
  if (i < np) o[i] = (i < n) ? a[i] : 0.0f;
}

__global__ void softmax_kernel(const float* __restrict__ S, bf16_t* __restrict__ P) {
  const int row = blockIdx.x;
  const float* r = S + (size_t)row * NS;
  __shared__ float red[256];
  const int tid = threadIdx.x;
  float m = -1e30f;
  for (int i = tid; i < NS; i += 256) m = fmaxf(m, r[i]);
  red[tid] = m; __syncthreads();
  for (int s = 128; s > 0; s >>= 1) { if (tid < s) red[tid] = fmaxf(red[tid], red[tid + s]); __syncthreads(); }
  const float mx = red[0]; __syncthreads();
  const float e0 = __expf(r[tid] - mx);
  const float e1 = __expf(r[tid + 256] - mx);
  red[tid] = e0 + e1; __syncthreads();
  for (int s = 128; s > 0; s >>= 1) { if (tid < s) red[tid] += red[tid + s]; __syncthreads(); }
  const float inv = 1.0f / red[0];
  bf16_t* p = P + (size_t)row * NS;
  p[tid]       = f2bf(e0 * inv);
  p[tid + 256] = f2bf(e1 * inv);
}

// CRF forward NLL. One block, 288 threads = 32 batches x 9 states.
__global__ void crf_kernel(const float* __restrict__ em /* [NBS, NTp] */,
                           const int* __restrict__ tags,
                           const unsigned char* __restrict__ mask,
                           const float* __restrict__ start,
                           const float* __restrict__ endv,
                           const float* __restrict__ trans,
                           float* __restrict__ out) {
  __shared__ float alphaS[NB][NT];
  __shared__ float denomS[NB];
  __shared__ float scoreS[NB];
  __shared__ float tS[NT * NT];
  __shared__ float red[NB];
  const int tid = threadIdx.x;
  if (tid < NT * NT) tS[tid] = trans[tid];
  const int b = tid / NT, j = tid % NT;
  if (tid < NB * NT) alphaS[b][j] = start[j] + em[(size_t)(b * NS) * NTp + j];
  __syncthreads();
  for (int t = 1; t < NS; ++t) {
    float nxt = 0.0f;
    if (tid < NB * NT) {
      float mx = -1e30f;
#pragma unroll
      for (int i = 0; i < NT; ++i) mx = fmaxf(mx, alphaS[b][i] + tS[i * NT + j]);
      float s = 0.0f;
#pragma unroll
      for (int i = 0; i < NT; ++i) s += __expf(alphaS[b][i] + tS[i * NT + j] - mx);
      const float nv = mx + __logf(s) + em[(size_t)(b * NS + t) * NTp + j];
      nxt = mask[b * NS + t] ? nv : alphaS[b][j];
    }
    __syncthreads();
    if (tid < NB * NT) alphaS[b][j] = nxt;
    __syncthreads();
  }
  if (tid < NB) {
    float mx = -1e30f;
    for (int i = 0; i < NT; ++i) mx = fmaxf(mx, alphaS[tid][i] + endv[i]);
    float s = 0.0f;
    for (int i = 0; i < NT; ++i) s += __expf(alphaS[tid][i] + endv[i] - mx);
    denomS[tid] = mx + __logf(s);
    const int* tg = tags + tid * NS;
    const unsigned char* mk = mask + tid * NS;
    float sc = start[tg[0]] + em[(size_t)(tid * NS) * NTp + tg[0]];
    int last = tg[0];
    for (int t = 1; t < NS; ++t) {
      if (mk[t]) {
        sc += tS[last * NT + tg[t]] + em[(size_t)(tid * NS + t) * NTp + tg[t]];
        last = tg[t];
      }
    }
    sc += endv[last];
    scoreS[tid] = sc;
  }
  __syncthreads();
  if (tid < NB) red[tid] = denomS[tid] - scoreS[tid];
  __syncthreads();
  if (tid == 0) {
    float s = 0.0f;
    for (int i = 0; i < NB; ++i) s += red[i];
    out[0] = s;
  }
}

// ---------------------------------------------------------------------------
// Workspace layout
// ---------------------------------------------------------------------------
constexpr size_t AL256(size_t x) { return (x + 255) & ~(size_t)255; }
constexpr size_t OFF_X0    = 0;                                                   // [NBS,320] bf16
constexpr size_t OFF_WIH1  = AL256(OFF_X0 + (size_t)NBS * NEp * 2);               // 2x [1024,320] bf16
constexpr size_t OFF_WHH   = AL256(OFF_WIH1 + 2 * (size_t)NG * NEp * 2);          // 4x [1024,256] bf16
constexpr size_t OFF_WIH2  = AL256(OFF_WHH + 4 * (size_t)NG * NH * 2);            // 2x [1024,512] bf16
constexpr size_t OFF_BIAS  = AL256(OFF_WIH2 + 2 * (size_t)NG * ND * 2);           // 4x [1024] f32
constexpr size_t OFF_WQKV  = AL256(OFF_BIAS + 4 * (size_t)NG * 4);                // 3x [512,512] bf16
constexpr size_t OFF_WEMIT = AL256(OFF_WQKV + 3 * (size_t)ND * ND * 2);           // [16,512] bf16
constexpr size_t OFF_BEMIT = AL256(OFF_WEMIT + (size_t)NTp * ND * 2);             // [16] f32
constexpr size_t OFF_XG    = AL256(OFF_BEMIT + NTp * 4);                          // 2x [NBS,1024] f32
constexpr size_t XG_BYTES  = 2 * (size_t)NBS * NG * 4;
// attention-phase buffers reuse the (dead) XG region:
constexpr size_t OFF_SCORES = OFF_XG;                                             // [32,512,512] f32
constexpr size_t OFF_ALPHA  = OFF_SCORES + (size_t)NB * NS * NS * 4;              // bf16
constexpr size_t OFF_ATT    = OFF_ALPHA + (size_t)NB * NS * NS * 2;               // [NBS,512] bf16
constexpr size_t OFF_EMIS   = OFF_ATT + (size_t)NBS * ND * 2;                     // [NBS,16] f32
constexpr size_t OFF_L1OUT  = AL256(OFF_XG + XG_BYTES);                           // [NBS,512] bf16
constexpr size_t OFF_L2OUT  = AL256(OFF_L1OUT + (size_t)NBS * ND * 2);
constexpr size_t OFF_Q      = AL256(OFF_L2OUT + (size_t)NBS * ND * 2);
constexpr size_t OFF_K      = AL256(OFF_Q + (size_t)NBS * ND * 2);
constexpr size_t OFF_V      = AL256(OFF_K + (size_t)NBS * ND * 2);

extern "C" void kernel_launch(void* const* d_in, const int* in_sizes, int n_in,
                              void* d_out, int out_size, void* d_ws, size_t ws_size,
                              hipStream_t stream) {
  (void)in_sizes; (void)n_in; (void)out_size; (void)ws_size;

  const int*           ids    = (const int*)d_in[0];
  const unsigned char* maskp  = (const unsigned char*)d_in[1];
  const int*           labels = (const int*)d_in[2];
  const float*         table  = (const float*)d_in[3];
  // lstm_params flatten order: (layer, dir) x (w_ih, w_hh, b_ih, b_hh)
  const float* w_ih[2][2] = {{(const float*)d_in[4],  (const float*)d_in[8]},
                             {(const float*)d_in[12], (const float*)d_in[16]}};
  const float* w_hh[2][2] = {{(const float*)d_in[5],  (const float*)d_in[9]},
                             {(const float*)d_in[13], (const float*)d_in[17]}};
  const float* b_ih[2][2] = {{(const float*)d_in[6],  (const float*)d_in[10]},
                             {(const float*)d_in[14], (const float*)d_in[18]}};
  const float* b_hh[2][2] = {{(const float*)d_in[7],  (const float*)d_in[11]},
                             {(const float*)d_in[15], (const float*)d_in[19]}};
  const float* wq = (const float*)d_in[20];
  const float* wk = (const float*)d_in[21];
  const float* wv = (const float*)d_in[22];
  const float* w_emit    = (const float*)d_in[23];
  const float* b_emit    = (const float*)d_in[24];
  const float* crf_start = (const float*)d_in[25];
  const float* crf_end   = (const float*)d_in[26];
  const float* crf_trans = (const float*)d_in[27];

  char* ws = (char*)d_ws;
  bf16_t* x0     = (bf16_t*)(ws + OFF_X0);
  bf16_t* wih1B  = (bf16_t*)(ws + OFF_WIH1);
  bf16_t* whhB   = (bf16_t*)(ws + OFF_WHH);
  bf16_t* wih2B  = (bf16_t*)(ws + OFF_WIH2);
  float*  biasB  = (float*)(ws + OFF_BIAS);
  bf16_t* wqkvB  = (bf16_t*)(ws + OFF_WQKV);
  bf16_t* wemitB = (bf16_t*)(ws + OFF_WEMIT);
  float*  bemitB = (float*)(ws + OFF_BEMIT);
  float*  xg[2]  = {(float*)(ws + OFF_XG), (float*)(ws + OFF_XG) + (size_t)NBS * NG};
  float*  scores = (float*)(ws + OFF_SCORES);
  bf16_t* alphaB = (bf16_t*)(ws + OFF_ALPHA);
  bf16_t* attB   = (bf16_t*)(ws + OFF_ATT);
  float*  emisF  = (float*)(ws + OFF_EMIS);
  bf16_t* l1out  = (bf16_t*)(ws + OFF_L1OUT);
  bf16_t* l2out  = (bf16_t*)(ws + OFF_L2OUT);
  bf16_t* qB     = (bf16_t*)(ws + OFF_Q);
  bf16_t* kB     = (bf16_t*)(ws + OFF_K);
  bf16_t* vB     = (bf16_t*)(ws + OFF_V);

  auto packpad = [&](const float* src, bf16_t* dst, int sr, int sc, int dr, int dc) {
    const int n = dr * dc;
    pack_pad_kernel<<<(n + 255) / 256, 256, 0, stream>>>(src, dst, sr, sc, dr, dc);
  };

  // --- 1. pack / convert parameters to bf16 (padded) -----------------------
  bf16_t* wih1P[2], *whhP[2][2], *wih2P[2];
  float*  biasP[2][2];
  for (int d = 0; d < 2; ++d) {
    wih1P[d] = wih1B + (size_t)d * NG * NEp;
    wih2P[d] = wih2B + (size_t)d * NG * ND;
    packpad(w_ih[0][d], wih1P[d], NG, NE, NG, NEp);
    packpad(w_ih[1][d], wih2P[d], NG, ND, NG, ND);
    for (int l = 0; l < 2; ++l) {
      whhP[l][d] = whhB + ((size_t)(l * 2 + d)) * NG * NH;
      biasP[l][d] = biasB + (size_t)(l * 2 + d) * NG;
      packpad(w_hh[l][d], whhP[l][d], NG, NH, NG, NH);
      add2_kernel<<<(NG + 255) / 256, 256, 0, stream>>>(b_ih[l][d], b_hh[l][d], biasP[l][d], NG);
    }
  }
  bf16_t* wqP = wqkvB;
  bf16_t* wkP = wqkvB + (size_t)ND * ND;
  bf16_t* wvP = wqkvB + 2 * (size_t)ND * ND;
  packpad(wq, wqP, ND, ND, ND, ND);
  packpad(wk, wkP, ND, ND, ND, ND);
  packpad(wv, wvP, ND, ND, ND, ND);
  packpad(w_emit, wemitB, NT, ND, NTp, ND);
  pad_vec_kernel<<<1, 32, 0, stream>>>(b_emit, bemitB, NT, NTp);

  // --- 2. embedding gather (bf16, K padded 300->320) -----------------------
  embed_gather_kernel<<<NBS, NEp, 0, stream>>>(ids, table, x0);

  // --- 3. layer 1: xg = x @ w_ih^T, then persistent WMMA scan --------------
  dim3 gXG(NG / 64, NBS / 32, 1);
  for (int d = 0; d < 2; ++d)
    gemm_kernel<0><<<gXG, 64, 0, stream>>>(x0, wih1P[d], xg[d], nullptr, nullptr,
                                           NBS, NG, NEp, NEp, NEp, NG, 0, 0, 0, 1.0f);
  const size_t ldsScan = 304 * 1024;  // h + c + gates + 2 weight slabs
  lstm_scan_kernel<<<2, 512, ldsScan, stream>>>(xg[0], xg[1], whhP[0][0], whhP[0][1],
                                                biasP[0][0], biasP[0][1], l1out);

  // --- 4. layer 2 ----------------------------------------------------------
  for (int d = 0; d < 2; ++d)
    gemm_kernel<0><<<gXG, 64, 0, stream>>>(l1out, wih2P[d], xg[d], nullptr, nullptr,
                                           NBS, NG, ND, ND, ND, NG, 0, 0, 0, 1.0f);
  lstm_scan_kernel<<<2, 512, ldsScan, stream>>>(xg[0], xg[1], whhP[1][0], whhP[1][1],
                                                biasP[1][0], biasP[1][1], l2out);

  // --- 5. attention --------------------------------------------------------
  dim3 gQKV(ND / 64, NBS / 32, 1);
  gemm_kernel<0><<<gQKV, 64, 0, stream>>>(l2out, wqP, nullptr, qB, nullptr,
                                          NBS, ND, ND, ND, ND, ND, 0, 0, 0, 1.0f);
  gemm_kernel<0><<<gQKV, 64, 0, stream>>>(l2out, wkP, nullptr, kB, nullptr,
                                          NBS, ND, ND, ND, ND, ND, 0, 0, 0, 1.0f);
  gemm_kernel<0><<<gQKV, 64, 0, stream>>>(l2out, wvP, nullptr, vB, nullptr,
                                          NBS, ND, ND, ND, ND, ND, 0, 0, 0, 1.0f);

  const float invSqrtD = 1.0f / sqrtf((float)(2 * NH));
  dim3 gAtt(NS / 64, NS / 32, NB);  // per-batch 512x512 tiles
  gemm_kernel<0><<<gAtt, 64, 0, stream>>>(qB, kB, scores, nullptr, nullptr,
                                          NS, NS, ND, ND, ND, NS,
                                          (long long)NS * ND, (long long)NS * ND,
                                          (long long)NS * NS, invSqrtD);
  softmax_kernel<<<NBS, 256, 0, stream>>>(scores, alphaB);
  gemm_kernel<1><<<gAtt, 64, 0, stream>>>(alphaB, vB, nullptr, attB, nullptr,
                                          NS, ND, NS, NS, ND, ND,
                                          (long long)NS * NS, (long long)NS * ND,
                                          (long long)NS * ND, 1.0f);

  // --- 6. emissions + CRF --------------------------------------------------
  dim3 gEm(1, NBS / 32, 1);
  gemm_kernel<0><<<gEm, 64, 0, stream>>>(attB, wemitB, emisF, nullptr, bemitB,
                                         NBS, NTp, ND, ND, ND, NTp, 0, 0, 0, 1.0f);
  crf_kernel<<<1, 288, 0, stream>>>(emisF, labels, maskp, crf_start, crf_end,
                                    crf_trans, (float*)d_out);
}
